// GRUNet_91250875171546
// MI455X (gfx1250) — matrix operand order repaired and checked by reading, not compiled, not run on hardware
//
#include <hip/hip_runtime.h>

typedef __attribute__((ext_vector_type(16))) _Float16 v16h;
typedef __attribute__((ext_vector_type(8)))  _Float16 v8h;
typedef __attribute__((ext_vector_type(8)))  float    v8f;

#define WAVES 8
#define PEDS_PER_BLK 128   // 8 waves * 16 peds

// fast sigmoid/tanh: v_exp_f32 + v_rcp_f32, no IEEE-div expansion
__device__ __forceinline__ float fsig(float x) {
    return __builtin_amdgcn_rcpf(1.0f + __expf(-x));
}
__device__ __forceinline__ float ftanh(float x) {
    return __builtin_fmaf(2.0f, __builtin_amdgcn_rcpf(1.0f + __expf(-2.0f * x)), -1.0f);
}

__device__ __forceinline__ v8f vbcast(float x) {
    v8f r;
#pragma unroll
    for (int i = 0; i < 8; ++i) r[i] = x;
    return r;
}

__global__ __launch_bounds__(256) void gru_traj_kernel(
    const float* __restrict__ obs,     // (8, peds, 2)
    const float* __restrict__ W_emb,   // (64, 2)
    const float* __restrict__ b_emb,   // (64)
    const float* __restrict__ W_ih,    // (192, 64)
    const float* __restrict__ b_ih,    // (192)
    const float* __restrict__ W_hh,    // (192, 64)
    const float* __restrict__ b_hh,    // (192)
    const float* __restrict__ W_out,   // (2, 64)
    const float* __restrict__ b_out,   // (2)
    float* __restrict__ out,           // (pred_len, peds, 2)
    int peds, int pred_len)
{
    // --- LDS ---
    __shared__ alignas(32) _Float16 sWhh[24 * 512];       // B fragments: frag(t,kc)*512 + lane*16 + i
    __shared__ alignas(32) _Float16 sH[WAVES][16][64];    // h, f16, row-major per wave tile
    __shared__ alignas(16) float    sX[WAVES][16][2];     // current input x (or fed-back o)
    __shared__ alignas(16) float    sMf[384];             // folded input matrix, interleaved pairs
    __shared__ alignas(16) float    sCf[192];             // folded input bias (+ b_hh for r,z)
    __shared__ alignas(16) float    sBhn[64];             // b_hh for n gate (must stay in gh)
    __shared__ alignas(16) float    sWout[128];
    __shared__ float                sBout[2];

    const int tid = threadIdx.x;

    // ---- per-block init: swizzle W_hh into WMMA-B fragment layout (f16) ----
    // B 32x16 tile (16x16x32 f16 wmma): lane l holds column n = 16t + (l&15),
    // halves i=0..15 cover K = 32*kc + 16*(l>>4) + i  (contiguous-K split, same order as the A fragments)
    for (int idx = tid; idx < 24 * 512; idx += 256) {
        int f = idx >> 9;           // fragment = t*2 + kc
        int r = idx & 511;
        int l = r >> 4;             // lane 0..31
        int i = r & 15;             // half 0..15
        int t = f >> 1, kc = f & 1;
        int n = (t << 4) + (l & 15);
        int K = (kc << 5) + ((l >> 4) << 4) + i;
        sWhh[idx] = (_Float16)W_hh[n * 64 + K];
    }
    // ---- fold input path: M = W_ih @ W_emb (192x2), c = W_ih @ b_emb + b_ih ----
    // r/z gates (j<128): fold b_hh additively into the constant as well.
    for (int j = tid; j < 192; j += 256) {
        float m0 = 0.f, m1 = 0.f, cc = 0.f;
#pragma unroll 8
        for (int k = 0; k < 64; ++k) {
            float wik = W_ih[j * 64 + k];
            m0 += wik * W_emb[k * 2 + 0];
            m1 += wik * W_emb[k * 2 + 1];
            cc += wik * b_emb[k];
        }
        sMf[2 * j + 0] = m0;
        sMf[2 * j + 1] = m1;
        sCf[j] = cc + b_ih[j] + (j < 128 ? b_hh[j] : 0.0f);
        if (j >= 128) sBhn[j - 128] = b_hh[j];
    }
    if (tid < 128) sWout[tid] = W_out[tid];
    if (tid < 2)   sBout[tid] = b_out[tid];
    // zero h in LDS (h0 = 0)
    for (int i = tid; i < 4096; i += 256) ((int*)sH)[i] = 0;
    __syncthreads();

    const int w    = tid >> 5;       // wave id 0..7
    const int l    = tid & 31;       // lane
    const int g    = l >> 4;         // lane half
    const int n16  = l & 15;
    const int ped0 = blockIdx.x * PEDS_PER_BLK + w * 16;
    const int p    = l >> 1;         // out-phase: ped within tile
    const int d    = l & 1;          //            output dim

    // ---- hoist all loop-invariant per-lane constants out of the step loop ----
    const v8f zeroC = vbcast(0.0f);
    v8f bnC[4];                      // n-gate b_hh broadcast, C operand for n-tile WMMA
    float2 mrv[4], mzv[4], mnv[4];
    float  crv[4], czv[4], cnv[4];
#pragma unroll
    for (int ct = 0; ct < 4; ++ct) {
        int jr = ct * 16 + n16, jz = jr + 64, jn = jr + 128;
        bnC[ct] = vbcast(sBhn[jr]);
        mrv[ct] = *(const float2*)&sMf[2 * jr]; crv[ct] = sCf[jr];
        mzv[ct] = *(const float2*)&sMf[2 * jz]; czv[ct] = sCf[jz];
        mnv[ct] = *(const float2*)&sMf[2 * jn]; cnv[ct] = sCf[jn];
    }

    v8f hreg[4];                     // h in f32 C-layout: tile ct, VGPR v -> (ped v+8g, col 16ct+n16)
#pragma unroll
    for (int i = 0; i < 4; ++i) hreg[i] = vbcast(0.0f);

    const int total_steps = 8 + pred_len;

    for (int step = 0; step < total_steps; ++step) {
        __syncthreads();   // prev step's sH/sX writes visible

        // ---- fetch x into sX (observed steps); pred steps use fed-back o ----
        if (step < 8) {
            float xv = 0.f;
            if (ped0 + p < peds)
                xv = obs[((size_t)step * peds + (ped0 + p)) * 2 + d];
            sX[w][p][d] = xv;
        }

        // ---- A fragments of h (16x32 f16, contiguous-K per lane half) from sH ----
        const _Float16* hrow = &sH[w][n16][0];
        v8h lo0 = *(const v8h*)(hrow + 16 * g);
        v8h hi0 = *(const v8h*)(hrow + 8 + 16 * g);
        v8h lo1 = *(const v8h*)(hrow + 32 + 16 * g);
        v8h hi1 = *(const v8h*)(hrow + 40 + 16 * g);
        v16h a0, a1;
#pragma unroll
        for (int i = 0; i < 8; ++i) {
            a0[i] = lo0[i]; a0[8 + i] = hi0[i];
            a1[i] = lo1[i]; a1[8 + i] = hi1[i];
        }

        __syncthreads();   // sX writes visible across lanes

        // x pairs for the 8 rows this lane accumulates (rows v+8g)
        float2 xr[8];
#pragma unroll
        for (int v = 0; v < 8; ++v)
            xr[v] = *(const float2*)&sX[w][v + 8 * g][0];

        // ---- per 16-column group: gh via WMMA, gi via folded VALU, gate math ----
#pragma unroll
        for (int ct = 0; ct < 4; ++ct) {
            v8f acc[3];
#pragma unroll
            for (int q = 0; q < 3; ++q) {
                int jt = ct + 4 * q;                 // r/z/n tile
                v8f c0 = (q < 2) ? zeroC : bnC[ct];  // loop-invariant C operand
                v16h b0 = *(const v16h*)&sWhh[(jt * 2 + 0) * 512 + l * 16];
                v16h b1 = *(const v16h*)&sWhh[(jt * 2 + 1) * 512 + l * 16];
                v8f c1 = __builtin_amdgcn_wmma_f32_16x16x32_f16(false, a0, false, b0,
                                                                (short)0, c0, false, false);
                acc[q] = __builtin_amdgcn_wmma_f32_16x16x32_f16(false, a1, false, b1,
                                                                (short)0, c1, false, false);
            }
            float2 mr = mrv[ct], mz = mzv[ct], mn = mnv[ct];
            float  cr = crv[ct], cz = czv[ct], cn = cnv[ct];

            v8f hnew;
#pragma unroll
            for (int v = 0; v < 8; ++v) {
                float x0 = xr[v].x, x1 = xr[v].y;
                float gir = cr + mr.x * x0 + mr.y * x1;
                float giz = cz + mz.x * x0 + mz.y * x1;
                float gin = cn + mn.x * x0 + mn.y * x1;
                float rg = fsig(gir + acc[0][v]);
                float zg = fsig(giz + acc[1][v]);
                float ng = ftanh(gin + rg * acc[2][v]);
                // (1-z)*n + z*h  ==  n + z*(h - n)
                hnew[v] = ng + zg * (hreg[ct][v] - ng);
            }
            hreg[ct] = hnew;
            // write h back to LDS (f16) for next step's A fragments / output proj
#pragma unroll
            for (int v = 0; v < 8; ++v)
                sH[w][v + 8 * g][ct * 16 + n16] = (_Float16)hnew[v];
        }

        __syncthreads();   // sH writes visible for out-projection

        // ---- output projection o = h @ W_out^T + b_out (steps 7..end) ----
        if (step >= 7) {
            const _Float16* hr = &sH[w][p][0];
            float o = sBout[d];
#pragma unroll
            for (int kk = 0; kk < 8; ++kk) {
                v8h hv = *(const v8h*)(hr + kk * 8);
#pragma unroll
                for (int i = 0; i < 8; ++i)
                    o += (float)hv[i] * sWout[d * 64 + kk * 8 + i];
            }
            if (step >= 8 && (ped0 + p) < peds)
                out[((size_t)(step - 8) * peds + (ped0 + p)) * 2 + d] = o;
            if (step < total_steps - 1)
                sX[w][p][d] = o;   // autoregressive feedback
        }
    }
}

extern "C" void kernel_launch(void* const* d_in, const int* in_sizes, int n_in,
                              void* d_out, int out_size, void* d_ws, size_t ws_size,
                              hipStream_t stream) {
    (void)n_in; (void)d_ws; (void)ws_size;
    const float* obs   = (const float*)d_in[0];
    const float* W_emb = (const float*)d_in[1];
    const float* b_emb = (const float*)d_in[2];
    const float* W_ih  = (const float*)d_in[3];
    const float* W_hh  = (const float*)d_in[4];
    const float* b_ih  = (const float*)d_in[5];
    const float* b_hh  = (const float*)d_in[6];
    const float* W_out = (const float*)d_in[7];
    const float* b_out = (const float*)d_in[8];
    float* out = (float*)d_out;

    int peds = in_sizes[0] / 16;                       // (seq=8, peds, 2)
    int pred_len = peds > 0 ? out_size / (2 * peds) : 12;
    int blocks = (peds + PEDS_PER_BLK - 1) / PEDS_PER_BLK;

    gru_traj_kernel<<<dim3(blocks), dim3(256), 0, stream>>>(
        obs, W_emb, b_emb, W_ih, b_ih, W_hh, b_hh, W_out, b_out,
        out, peds, pred_len);
}
